// Inter_pos_Update_62672162783746
// MI455X (gfx1250) — matrix operand order repaired and checked by reading, not compiled, not run on hardware
//
#include <hip/hip_runtime.h>
#include <hip/hip_bf16.h>
#include <math.h>

#define NN   50000
#define EE   800000
#define HIDD 128
#define EDIMM 64
#define DDIMM 64
#define TDIMM 128

typedef __attribute__((ext_vector_type(16))) _Float16 v16h;
typedef __attribute__((ext_vector_type(8)))  _Float16 v8h;
typedef __attribute__((ext_vector_type(8)))  float    v8f;

// ---------------------------------------------------------------------------
// WMMA helpers (CDNA5 wave32, V_WMMA_F32_16X16X32_F16)
// ---------------------------------------------------------------------------
__device__ __forceinline__ v8f wmma16(v16h a, v16h b, v8f c) {
  return __builtin_amdgcn_wmma_f32_16x16x32_f16(
      /*neg_a=*/false, a, /*neg_b=*/false, b,
      /*c_mod=*/(short)0, c, /*reuse_a=*/false, /*reuse_b=*/false);
}

// A-matrix fragment (16xK row-major f16 in LDS/global, 32-wide K-slab at k0).
// Lane L: m = L&15 ; halves 0..7 -> K = k0 + 8*(L>>4) + 0..7
//                    halves 8..15 -> K = k0 + 16 + 8*(L>>4) + 0..7
__device__ __forceinline__ v16h load_a_frag(const _Float16* base, int ld,
                                            int k0, int lane) {
  const int m  = lane & 15;
  const int kb = (lane >> 4) << 3;
  const _Float16* p = base + m * ld + k0 + kb;
  v8h lo = *(const v8h*)(p);
  v8h hi = *(const v8h*)(p + 16);
  v16h a;
#pragma unroll
  for (int j = 0; j < 8; ++j) { a[j] = lo[j]; a[j + 8] = hi[j]; }
  return a;
}

// B-matrix fragment (32x16 slice of KxNall row-major f16 weights):
// lane L holds column n0 + (L&15); halves 0..7 -> K = k0 + 8*(L>>4) + 0..7,
// halves 8..15 -> K = k0 + 16 + 8*(L>>4) + 0..7.
__device__ __forceinline__ v16h load_b_frag(const _Float16* W, int ldw,
                                            int k0, int n0, int lane) {
  const int n  = n0 + (lane & 15);
  const int kb = k0 + ((lane >> 4) << 3);
  v16h b;
#pragma unroll
  for (int j = 0; j < 8; ++j) {
    b[j]     = W[(size_t)(kb + j) * ldw + n];
    b[j + 8] = W[(size_t)(kb + 16 + j) * ldw + n];
  }
  return b;
}

// A fragment built from a row of f32 data in global memory (rowp = row base).
__device__ __forceinline__ v16h f32_frag(const float* __restrict__ rowp,
                                         int k0, int lane) {
  const int kb = (lane >> 4) << 3;
  const float4* p0 = (const float4*)(rowp + k0 + kb);
  const float4* p1 = (const float4*)(rowp + k0 + 16 + kb);
  float4 x0 = p0[0], x1 = p0[1], y0 = p1[0], y1 = p1[1];
  v16h a;
  a[0] = (_Float16)x0.x; a[1] = (_Float16)x0.y;
  a[2] = (_Float16)x0.z; a[3] = (_Float16)x0.w;
  a[4] = (_Float16)x1.x; a[5] = (_Float16)x1.y;
  a[6] = (_Float16)x1.z; a[7] = (_Float16)x1.w;
  a[8]  = (_Float16)y0.x; a[9]  = (_Float16)y0.y;
  a[10] = (_Float16)y0.z; a[11] = (_Float16)y0.w;
  a[12] = (_Float16)y1.x; a[13] = (_Float16)y1.y;
  a[14] = (_Float16)y1.z; a[15] = (_Float16)y1.w;
  return a;
}

// Gaussian-smearing A fragment computed directly into registers.
// Columns of the [attr|gauss] matrix >= 64 hold exp(coeff*(d - step*g)^2).
__device__ __forceinline__ v16h gauss_frag(float d, int k0, int lane) {
  const float step  = 15.0f / 63.0f;
  const float coeff = -0.5f / (step * step);
  const int kb = (lane >> 4) << 3;
  const int g0 = (k0 - EDIMM) + kb;
  v16h a;
#pragma unroll
  for (int j = 0; j < 8; ++j) {
    float t = d - step * (float)(g0 + j);
    a[j] = (_Float16)__expf(coeff * t * t);
  }
#pragma unroll
  for (int j = 0; j < 8; ++j) {
    float t = d - step * (float)(g0 + 16 + j);
    a[j + 8] = (_Float16)__expf(coeff * t * t);
  }
  return a;
}

// hl[row]*hr[col] Hadamard fused into the A-fragment gather (f16 vector mul).
__device__ __forceinline__ v16h hf_frag(const _Float16* __restrict__ hlrow,
                                        const _Float16* __restrict__ hrrow,
                                        int k0, int lane) {
  const int kb = (lane >> 4) << 3;
  v8h a0 = *(const v8h*)(hlrow + k0 + kb) * *(const v8h*)(hrrow + k0 + kb);
  v8h a1 = *(const v8h*)(hlrow + k0 + 16 + kb) *
           *(const v8h*)(hrrow + k0 + 16 + kb);
  v16h a;
#pragma unroll
  for (int j = 0; j < 8; ++j) { a[j] = a0[j]; a[j + 8] = a1[j]; }
  return a;
}

// fast silu: x * v_rcp_f32(1+exp(-x))   (f16-pipeline precision is fine here)
__device__ __forceinline__ float silu_f(float x) {
  return x * __builtin_amdgcn_rcpf(1.0f + __expf(-x));
}

// ---------------------------------------------------------------------------
// Prep kernels
// ---------------------------------------------------------------------------
__global__ void k_cvt(const float* __restrict__ src, _Float16* __restrict__ dst,
                      int n) {
  for (int i = blockIdx.x * blockDim.x + threadIdx.x; i < n;
       i += gridDim.x * blockDim.x)
    dst[i] = (_Float16)src[i];
}

__global__ void k_zero(float* __restrict__ p, int n) {
  for (int i = blockIdx.x * blockDim.x + threadIdx.x; i < n;
       i += gridDim.x * blockDim.x)
    p[i] = 0.0f;
}

// ---------------------------------------------------------------------------
// Per-node MLPs: hl = silu(h@w1l+b1l)@w2l+b2l ; hr likewise.  One wave per
// 16-node tile; both layers via v_wmma_f32_16x16x32_f16.
// ---------------------------------------------------------------------------
__global__ void __launch_bounds__(32) k_node_mlp(
    const float* __restrict__ h,
    const _Float16* __restrict__ w1l, const float* __restrict__ b1l,
    const _Float16* __restrict__ w2l, const float* __restrict__ b2l,
    const _Float16* __restrict__ w1r, const float* __restrict__ b1r,
    const _Float16* __restrict__ w2r, const float* __restrict__ b2r,
    _Float16* __restrict__ hl, _Float16* __restrict__ hr) {
  __shared__ __attribute__((aligned(32))) _Float16 sH[16 * HIDD];
  __shared__ __attribute__((aligned(32))) _Float16 sT[16 * 2 * HIDD];
  const int lane = threadIdx.x;
  const int base = blockIdx.x * 16;

  for (int i = lane; i < 16 * HIDD; i += 32) {
    int m = i >> 7, k = i & 127;
    sH[i] = (_Float16)h[(size_t)(base + m) * HIDD + k];
  }
  __syncthreads();

  v16h aH[4];
#pragma unroll
  for (int s = 0; s < 4; ++s) aH[s] = load_a_frag(sH, HIDD, 32 * s, lane);

  const int mrow = (lane >> 4) << 3;
  const int ncol = lane & 15;

  for (int pass = 0; pass < 2; ++pass) {
    const _Float16* w1 = pass ? w1r : w1l;
    const float*    b1 = pass ? b1r : b1l;
    const _Float16* w2 = pass ? w2r : w2l;
    const float*    b2 = pass ? b2r : b2l;
    _Float16* outp = pass ? hr : hl;

    for (int n0 = 0; n0 < 2 * HIDD; n0 += 16) {
      v8f c = {};
#pragma unroll
      for (int s = 0; s < 4; ++s)
        c = wmma16(aH[s], load_b_frag(w1, 2 * HIDD, 32 * s, n0, lane), c);
      float bias = b1[n0 + ncol];
#pragma unroll
      for (int i = 0; i < 8; ++i)
        sT[(i + mrow) * (2 * HIDD) + n0 + ncol] = (_Float16)silu_f(c[i] + bias);
    }
    __syncthreads();

    v16h aT[8];
#pragma unroll
    for (int s = 0; s < 8; ++s) aT[s] = load_a_frag(sT, 2 * HIDD, 32 * s, lane);
    for (int n0 = 0; n0 < HIDD; n0 += 16) {
      v8f c = {};
#pragma unroll
      for (int s = 0; s < 8; ++s)
        c = wmma16(aT[s], load_b_frag(w2, HIDD, 32 * s, n0, lane), c);
      float bias = b2[n0 + ncol];
#pragma unroll
      for (int i = 0; i < 8; ++i)
        outp[(size_t)(base + i + mrow) * HIDD + n0 + ncol] =
            (_Float16)(c[i] + bias);
    }
    __syncthreads();
  }
}

// ---------------------------------------------------------------------------
// Fused per-edge pipeline: one wave per 32-edge tile (two 16-row A tiles
// sharing every B fragment -> halves weight fragment traffic from L2).
// All E-sized intermediates stay in registers/LDS; atomics scatter forces.
// ---------------------------------------------------------------------------
__global__ void __launch_bounds__(32) k_edge(
    const float* __restrict__ pos, const int* __restrict__ eidx,
    const float* __restrict__ eattr, const float* __restrict__ timef,
    const _Float16* __restrict__ hl, const _Float16* __restrict__ hr,
    const _Float16* __restrict__ elw, const float* __restrict__ elb,
    const _Float16* __restrict__ ndw, const float* __restrict__ ndb,
    const _Float16* __restrict__ ifw1, const float* __restrict__ ifb1,
    const float* __restrict__ ifw2, const float* __restrict__ ifb2,
    float* __restrict__ dp) {
  __shared__ __attribute__((aligned(32))) _Float16 sZ [32 * 128];
  __shared__ __attribute__((aligned(32))) _Float16 sT1[32 * 256];
  __shared__ int   sRow[32];
  __shared__ int   sCol[32];
  __shared__ float sDist[32];
  __shared__ float sDir[32 * 3];

  const int lane = threadIdx.x;
  const long e0 = (long)blockIdx.x * 32;

  // per-edge scalars: one edge per lane, no divergence
  {
    const long e = e0 + lane;
    const int r = eidx[e];
    const int c = eidx[(long)EE + e];
    sRow[lane] = r;
    sCol[lane] = c;
    float dx = pos[r * 3 + 0] - pos[c * 3 + 0];
    float dy = pos[r * 3 + 1] - pos[c * 3 + 1];
    float dz = pos[r * 3 + 2] - pos[c * 3 + 2];
    float d = sqrtf(dx * dx + dy * dy + dz * dz);
    sDist[lane] = d;
    float inv = __builtin_amdgcn_rcpf(fmaxf(d, 1e-12f));
    sDir[lane * 3 + 0] = dx * inv;
    sDir[lane * 3 + 1] = dy * inv;
    sDir[lane * 3 + 2] = dz * inv;
  }
  __syncthreads();

  const int ml   = lane & 15;
  const int mrow = (lane >> 4) << 3;
  const int ncol = ml;

  // ---- hoist GEMM1/2 A fragments for both 16-row tiles --------------------
  v16h aE0[4], aE1[4], aH0[4], aH1[4];
  {
    const float* r0 = eattr + (size_t)(e0 + ml) * EDIMM;
    const float* r1 = eattr + (size_t)(e0 + 16 + ml) * EDIMM;
    aE0[0] = f32_frag(r0, 0, lane);  aE0[1] = f32_frag(r0, 32, lane);
    aE1[0] = f32_frag(r1, 0, lane);  aE1[1] = f32_frag(r1, 32, lane);
    float d0 = sDist[ml], d1 = sDist[16 + ml];
    aE0[2] = gauss_frag(d0, 64, lane);  aE0[3] = gauss_frag(d0, 96, lane);
    aE1[2] = gauss_frag(d1, 64, lane);  aE1[3] = gauss_frag(d1, 96, lane);

    const _Float16* hl0 = hl + (size_t)sRow[ml] * HIDD;
    const _Float16* hr0 = hr + (size_t)sCol[ml] * HIDD;
    const _Float16* hl1 = hl + (size_t)sRow[16 + ml] * HIDD;
    const _Float16* hr1 = hr + (size_t)sCol[16 + ml] * HIDD;
#pragma unroll
    for (int s = 0; s < 4; ++s) {
      aH0[s] = hf_frag(hl0, hr0, 32 * s, lane);
      aH1[s] = hf_frag(hl1, hr1, 32 * s, lane);
    }
  }

  // ---- e = [attr|gauss]@el_w ; nf = hf@nd_w ; z = (e+eb)*(nf+nb) ----------
  for (int n0 = 0; n0 < HIDD; n0 += 16) {
    v8f ce0 = {}, ce1 = {}, cn0 = {}, cn1 = {};
#pragma unroll
    for (int s = 0; s < 4; ++s) {
      v16h b = load_b_frag(elw, HIDD, 32 * s, n0, lane);
      ce0 = wmma16(aE0[s], b, ce0);
      ce1 = wmma16(aE1[s], b, ce1);
    }
#pragma unroll
    for (int s = 0; s < 4; ++s) {
      v16h b = load_b_frag(ndw, HIDD, 32 * s, n0, lane);
      cn0 = wmma16(aH0[s], b, cn0);
      cn1 = wmma16(aH1[s], b, cn1);
    }
    float be = elb[n0 + ncol], bn = ndb[n0 + ncol];
#pragma unroll
    for (int i = 0; i < 8; ++i) {
      sZ[(i + mrow) * 128 + n0 + ncol] =
          (_Float16)((ce0[i] + be) * (cn0[i] + bn));
      sZ[(16 + i + mrow) * 128 + n0 + ncol] =
          (_Float16)((ce1[i] + be) * (cn1[i] + bn));
    }
  }
  __syncthreads();

  // ---- t1 = silu([z|time] @ if_w1 + if_b1)   (K = 256) --------------------
  v16h aZ0[4], aZ1[4], aT0[4], aT1[4];
#pragma unroll
  for (int s = 0; s < 4; ++s) {
    aZ0[s] = load_a_frag(sZ, 128, 32 * s, lane);
    aZ1[s] = load_a_frag(sZ + 16 * 128, 128, 32 * s, lane);
  }
  {
    const float* t0 = timef + (size_t)(e0 + ml) * TDIMM;
    const float* t1 = timef + (size_t)(e0 + 16 + ml) * TDIMM;
#pragma unroll
    for (int s = 0; s < 4; ++s) {
      aT0[s] = f32_frag(t0, 32 * s, lane);
      aT1[s] = f32_frag(t1, 32 * s, lane);
    }
  }
  for (int n0 = 0; n0 < 2 * HIDD; n0 += 16) {
    v8f c0 = {}, c1 = {};
#pragma unroll
    for (int s = 0; s < 4; ++s) {
      v16h b = load_b_frag(ifw1, 2 * HIDD, 32 * s, n0, lane);
      c0 = wmma16(aZ0[s], b, c0);
      c1 = wmma16(aZ1[s], b, c1);
    }
#pragma unroll
    for (int s = 0; s < 4; ++s) {
      v16h b = load_b_frag(ifw1, 2 * HIDD, 128 + 32 * s, n0, lane);
      c0 = wmma16(aT0[s], b, c0);
      c1 = wmma16(aT1[s], b, c1);
    }
    float bv = ifb1[n0 + ncol];
#pragma unroll
    for (int i = 0; i < 8; ++i) {
      sT1[(i + mrow) * 256 + n0 + ncol] = (_Float16)silu_f(c0[i] + bv);
      sT1[(16 + i + mrow) * 256 + n0 + ncol] = (_Float16)silu_f(c1[i] + bv);
    }
  }
  __syncthreads();

  // ---- inter = t1 @ if_w2 + if_b2 ; force scatter (one edge per lane) -----
  {
    float acc = ifb2[0];
    const _Float16* t = sT1 + lane * 256;
#pragma unroll 8
    for (int k = 0; k < 256; ++k) acc += (float)t[k] * ifw2[k];
    float sc = acc * __builtin_amdgcn_rcpf(sDist[lane] + 1.0f);
    const int r = sRow[lane];
    atomicAdd(&dp[r * 3 + 0], sc * sDir[lane * 3 + 0]);
    atomicAdd(&dp[r * 3 + 1], sc * sDir[lane * 3 + 1]);
    atomicAdd(&dp[r * 3 + 2], sc * sDir[lane * 3 + 2]);
  }
}

// ---------------------------------------------------------------------------
// Final per-node scale: out = delta_pos * softplus(silu([h|nrm]@w1+b1)@w2+b2)
// ---------------------------------------------------------------------------
__global__ void __launch_bounds__(256) k_scale(
    const float* __restrict__ h, const float* __restrict__ dp,
    const float* __restrict__ w1, const float* __restrict__ b1,
    const float* __restrict__ w2, const float* __restrict__ b2,
    float* __restrict__ out) {
  const int lane = threadIdx.x & 31;
  const int wave = threadIdx.x >> 5;
  const int n = blockIdx.x * 8 + wave;
  if (n >= NN) return;
  float d0 = dp[n * 3 + 0], d1 = dp[n * 3 + 1], d2 = dp[n * 3 + 2];
  float nrm = sqrtf(d0 * d0 + d1 * d1 + d2 * d2);
  const int j0 = lane, j1 = lane + 32;
  float a0 = b1[j0], a1 = b1[j1];
#pragma unroll 4
  for (int k = 0; k < HIDD; ++k) {
    float s = h[(size_t)n * HIDD + k];
    a0 += s * w1[k * 64 + j0];
    a1 += s * w1[k * 64 + j1];
  }
  a0 += nrm * w1[HIDD * 64 + j0];
  a1 += nrm * w1[HIDD * 64 + j1];
  float part = silu_f(a0) * w2[j0] + silu_f(a1) * w2[j1];
#pragma unroll
  for (int off = 16; off > 0; off >>= 1) part += __shfl_xor(part, off, 32);
  float x = part + b2[0];
  float sc = (x > 20.0f) ? x : log1pf(__expf(x));
  if (lane == 0) {
    out[n * 3 + 0] = d0 * sc;
    out[n * 3 + 1] = d1 * sc;
    out[n * 3 + 2] = d2 * sc;
  }
}

// ---------------------------------------------------------------------------
extern "C" void kernel_launch(void* const* d_in, const int* in_sizes, int n_in,
                              void* d_out, int out_size, void* d_ws,
                              size_t ws_size, hipStream_t stream) {
  (void)in_sizes; (void)n_in; (void)out_size; (void)ws_size;
  const float* pos   = (const float*)d_in[0];
  const float* h     = (const float*)d_in[1];
  const int*   eidx  = (const int*)d_in[2];
  const float* eattr = (const float*)d_in[3];
  const float* timef = (const float*)d_in[4];
  const float* nl_w1 = (const float*)d_in[5];
  const float* nl_b1 = (const float*)d_in[6];
  const float* nl_w2 = (const float*)d_in[7];
  const float* nl_b2 = (const float*)d_in[8];
  const float* nr_w1 = (const float*)d_in[9];
  const float* nr_b1 = (const float*)d_in[10];
  const float* nr_w2 = (const float*)d_in[11];
  const float* nr_b2 = (const float*)d_in[12];
  const float* el_w  = (const float*)d_in[13];
  const float* el_b  = (const float*)d_in[14];
  const float* nd_w  = (const float*)d_in[15];
  const float* nd_b  = (const float*)d_in[16];
  const float* if_w1 = (const float*)d_in[17];
  const float* if_b1 = (const float*)d_in[18];
  const float* if_w2 = (const float*)d_in[19];
  const float* if_b2 = (const float*)d_in[20];
  const float* sc_w1 = (const float*)d_in[21];
  const float* sc_b1 = (const float*)d_in[22];
  const float* sc_w2 = (const float*)d_in[23];
  const float* sc_b2 = (const float*)d_in[24];

  _Float16* hl    = (_Float16*)d_ws;                 // N*128
  _Float16* hrp   = hl + (size_t)NN * HIDD;          // N*128
  _Float16* f_nl1 = hrp + (size_t)NN * HIDD;         // 128*256
  _Float16* f_nl2 = f_nl1 + 128 * 256;               // 256*128
  _Float16* f_nr1 = f_nl2 + 256 * 128;               // 128*256
  _Float16* f_nr2 = f_nr1 + 128 * 256;               // 256*128
  _Float16* f_elw = f_nr2 + 256 * 128;               // 128*128
  _Float16* f_ndw = f_elw + 128 * 128;               // 128*128
  _Float16* f_if1 = f_ndw + 128 * 128;               // 256*256
  float*    dp    = (float*)(f_if1 + 256 * 256);     // N*3 f32

  k_cvt<<<64, 256, 0, stream>>>(nl_w1, f_nl1, 128 * 256);
  k_cvt<<<64, 256, 0, stream>>>(nl_w2, f_nl2, 256 * 128);
  k_cvt<<<64, 256, 0, stream>>>(nr_w1, f_nr1, 128 * 256);
  k_cvt<<<64, 256, 0, stream>>>(nr_w2, f_nr2, 256 * 128);
  k_cvt<<<64, 256, 0, stream>>>(el_w,  f_elw, 128 * 128);
  k_cvt<<<64, 256, 0, stream>>>(nd_w,  f_ndw, 128 * 128);
  k_cvt<<<64, 256, 0, stream>>>(if_w1, f_if1, 256 * 256);
  k_zero<<<128, 256, 0, stream>>>(dp, NN * 3);

  k_node_mlp<<<NN / 16, 32, 0, stream>>>(h, f_nl1, nl_b1, f_nl2, nl_b2,
                                         f_nr1, nr_b1, f_nr2, nr_b2, hl, hrp);

  k_edge<<<EE / 32, 32, 0, stream>>>(pos, eidx, eattr, timef, hl, hrp,
                                     f_elw, el_b, f_ndw, nd_b,
                                     f_if1, if_b1, if_w2, if_b2, dp);

  k_scale<<<NN / 8, 256, 0, stream>>>(h, dp, sc_w1, sc_b1, sc_w2, sc_b2,
                                      (float*)d_out);
}
